// TimeDistributedCNN_29480655519842
// MI455X (gfx1250) — matrix operand order repaired and checked by reading, not compiled, not run on hardware
//
#include <hip/hip_runtime.h>
#include <hip/hip_bf16.h>
#include <math.h>

// Harness contract: kernel_launch only, all launches on `stream`, scratch in d_ws.

typedef _Float16 half16 __attribute__((ext_vector_type(16)));
typedef float    float8 __attribute__((ext_vector_type(8)));

#define EPSV 1e-5f

__device__ __forceinline__ float sigf(float x) { return 1.0f / (1.0f + __expf(-x)); }

// A-fragment (16x32 f16 tile, M x K) from a row-major f16 LDS array.
// Contiguous k order: lanes 0-15 -> row M=lane, K[0..15];
// lanes 16-31 -> row M=lane-16, K[16..31].
__device__ __forceinline__ half16 frag_a_lds(const _Float16* base, int rowStride,
                                             int m0, int kbase, int lane) {
  const int laneHi = (lane >> 4) & 1;
  const int row = m0 + (lane & 15);
  const _Float16* p = base + row * rowStride + kbase;
  const int o1 = laneHi ? 16 : 0;
  const int o2 = laneHi ? 24 : 8;
  half16 a;
#pragma unroll
  for (int j = 0; j < 8; ++j) a[j] = p[o1 + j];
#pragma unroll
  for (int j = 0; j < 8; ++j) a[8 + j] = p[o2 + j];
  return a;
}

// B-fragment (32x16 f16 tile, K x N), storage transposed [n][k] (contiguous k).
// lanes 0-15 -> col n, halves = K[0..15]; lanes 16-31 -> col n, halves = K[16..31].
// `rowp` must already point at the start of this lane's column (k index 0).
__device__ __forceinline__ half16 frag_b_lds_contig(const _Float16* rowp, int kbase, int lane) {
  const int laneHi = (lane >> 4) & 1;
  const _Float16* p = rowp + kbase + (laneHi ? 16 : 0);
  half16 bfr;
#pragma unroll
  for (int j = 0; j < 16; ++j) bfr[j] = p[j];
  return bfr;
}

// ---------------------------------------------------------------------------
// Kernel 1: fold BN into conv weights/bias (pre-packed per-tap layouts);
// precompute LSTM bias sums.
//   w2e layout: [kk(7)][oc(32)][ic(64)]   (f16)
//   w3e layout: [kk(5)][oc(16)][ic(32)]   (f16)
// ---------------------------------------------------------------------------
__global__ __launch_bounds__(256) void prep_weights(
    const float* __restrict__ c1w, const float* __restrict__ c1b,
    const float* __restrict__ g1, const float* __restrict__ be1,
    const float* __restrict__ m1, const float* __restrict__ v1,
    const float* __restrict__ c2w, const float* __restrict__ c2b,
    const float* __restrict__ g2, const float* __restrict__ be2,
    const float* __restrict__ m2, const float* __restrict__ v2,
    const float* __restrict__ c3w, const float* __restrict__ c3b,
    const float* __restrict__ g3, const float* __restrict__ be3,
    const float* __restrict__ m3, const float* __restrict__ v3,
    const float* __restrict__ bih0, const float* __restrict__ bhh0,
    const float* __restrict__ bih1, const float* __restrict__ bhh1,
    float* __restrict__ w1e, float* __restrict__ b1e,
    _Float16* __restrict__ w2e, float* __restrict__ b2e,
    _Float16* __restrict__ w3e, float* __restrict__ b3e,
    float* __restrict__ bs0, float* __restrict__ bs1) {
  const int tid = threadIdx.x;
  for (int i = tid; i < 64 * 9; i += 256) {
    int oc = i / 9; float s = g1[oc] * rsqrtf(v1[oc] + EPSV);
    w1e[i] = c1w[i] * s;
  }
  for (int i = tid; i < 64; i += 256) {
    float s = g1[i] * rsqrtf(v1[i] + EPSV);
    b1e[i] = (c1b[i] - m1[i]) * s + be1[i];
  }
  for (int i = tid; i < 7 * 32 * 64; i += 256) {
    int kk = i >> 11;            // /2048
    int rem = i & 2047;
    int oc = rem >> 6;
    int ic = rem & 63;
    float s = g2[oc] * rsqrtf(v2[oc] + EPSV);
    w2e[i] = (_Float16)(c2w[oc * 448 + ic * 7 + kk] * s);
  }
  for (int i = tid; i < 32; i += 256) {
    float s = g2[i] * rsqrtf(v2[i] + EPSV);
    b2e[i] = (c2b[i] - m2[i]) * s + be2[i];
  }
  for (int i = tid; i < 5 * 16 * 32; i += 256) {
    int kk = i >> 9;             // /512
    int rem = i & 511;
    int oc = rem >> 5;
    int ic = rem & 31;
    float s = g3[oc] * rsqrtf(v3[oc] + EPSV);
    w3e[i] = (_Float16)(c3w[oc * 160 + ic * 5 + kk] * s);
  }
  for (int i = tid; i < 16; i += 256) {
    float s = g3[i] * rsqrtf(v3[i] + EPSV);
    b3e[i] = (c3b[i] - m3[i]) * s + be3[i];
  }
  for (int i = tid; i < 256; i += 256) { bs0[i] = bih0[i] + bhh0[i]; bs1[i] = bih1[i] + bhh1[i]; }
}

// ---------------------------------------------------------------------------
// Kernel 2: fused per-window conv stack. One workgroup (4 waves) per window.
// conv1 in VALU; conv2/conv3 as tap-decomposed WMMA GEMMs with transposed,
// zero-guarded activations in LDS so every fragment is a contiguous
// ds_load_b128 (no predication, no divisions). mean -> feats[t*16+b][16].
// ---------------------------------------------------------------------------
#define H1T_ROWS 70   // 3 left pad + 50 valid + tail pad (p up to 63, kk up to 6)
#define H2T_ROWS 68   // 2 left pad + rows up to p=63,kk=4
__global__ __launch_bounds__(128) void conv_fused(
    const float* __restrict__ x, const float* __restrict__ w1e,
    const float* __restrict__ b1e, const _Float16* __restrict__ w2e,
    const float* __restrict__ b2e, const _Float16* __restrict__ w3e,
    const float* __restrict__ b3e, float* __restrict__ feats) {
  __shared__ float    s_win[64];                       // window padded by 4/10 zeros
  __shared__ float    s_w1[64 * 9];
  __shared__ float    s_b1[64];
  __shared__ __align__(16) _Float16 s_h1t[H1T_ROWS * 64];  // [pos+3][ic]
  __shared__ __align__(16) _Float16 s_w2[7 * 32 * 64];     // [kk][oc][ic]
  __shared__ float    s_b2[32];
  __shared__ __align__(16) _Float16 s_h2t[H2T_ROWS * 32];  // [pos+2][ic]
  __shared__ __align__(16) _Float16 s_w3[5 * 16 * 32];     // [kk][oc][ic]
  __shared__ float    s_b3[16];
  __shared__ float    s_sum[16];

  const int tid = threadIdx.x;
  const int wid = blockIdx.x;
  const int b = wid >> 11;       // 2048 windows per batch row
  const int t = wid & 2047;

  // ---- stage: window (+halo zeros), folded weights, zero guard rows ----
  for (int j = tid; j < 64; j += 128) {
    int p = j - 4;                              // window position
    int src = t + p - 49;                       // left zero-pad of W-1 in time
    s_win[j] = (p >= 0 && p < 50 && src >= 0) ? x[b * 2048 + src] : 0.0f;
  }
  for (int i = tid; i < 64 * 9; i += 128) s_w1[i] = w1e[i];
  for (int i = tid; i < 64; i += 128) s_b1[i] = b1e[i];
  {
    const uint32_t* src2 = (const uint32_t*)w2e;
    uint32_t* dst2 = (uint32_t*)s_w2;
    for (int i = tid; i < 7 * 32 * 64 / 2; i += 128) dst2[i] = src2[i];
    const uint32_t* src3 = (const uint32_t*)w3e;
    uint32_t* dst3 = (uint32_t*)s_w3;
    for (int i = tid; i < 5 * 16 * 32 / 2; i += 128) dst3[i] = src3[i];
    uint32_t* h1z = (uint32_t*)s_h1t;
    for (int i = tid; i < H1T_ROWS * 64 / 2; i += 128) h1z[i] = 0u;
    uint32_t* h2z = (uint32_t*)s_h2t;
    for (int i = tid; i < H2T_ROWS * 32 / 2; i += 128) h2z[i] = 0u;
  }
  for (int i = tid; i < 32; i += 128) s_b2[i] = b2e[i];
  for (int i = tid; i < 16; i += 128) { s_b3[i] = b3e[i]; s_sum[i] = 0.0f; }
  __syncthreads();

  // ---- conv1 (K=9, pad 4) + BN + ReLU in VALU, store transposed ----
  for (int i = tid; i < 64 * 50; i += 128) {
    int oc = i / 50, p = i - oc * 50;
    float acc = s_b1[oc];
#pragma unroll
    for (int k = 0; k < 9; ++k) acc += s_w1[oc * 9 + k] * s_win[p + k];  // s_win pre-shifted by 4
    s_h1t[(p + 3) * 64 + oc] = (_Float16)fmaxf(acc, 0.0f);
  }
  __syncthreads();

  const int wave = tid >> 5;
  const int lane = tid & 31;
  const int laneHi = lane >> 4;
  const int lx = lane & 15;
  const int n0 = wave * 16;      // output-position tile (cols 0..63, valid <50)
  const int p = n0 + lx;

  // ---- conv2: sum_kk W2[:,:,kk](32x64) @ h1[:, p+kk-3](64x50), WMMA ----
  float8 acc2[2];
#pragma unroll
  for (int mt = 0; mt < 2; ++mt)
#pragma unroll
    for (int r = 0; r < 8; ++r) acc2[mt][r] = s_b2[mt * 16 + (laneHi ? 8 : 0) + r];

#pragma unroll
  for (int kk = 0; kk < 7; ++kk) {
    const _Float16* brow = s_h1t + (p + kk) * 64;     // (p+kk-3)+3 pad rows
#pragma unroll
    for (int ks = 0; ks < 2; ++ks) {
      const int kb = ks * 32;
      half16 bfrag = frag_b_lds_contig(brow, kb, lane);
#pragma unroll
      for (int mt = 0; mt < 2; ++mt) {
        half16 afrag = frag_a_lds(s_w2 + kk * 2048, 64, mt * 16, kb, lane);
        acc2[mt] = __builtin_amdgcn_wmma_f32_16x16x32_f16(
            false, afrag, false, bfrag, (short)0, acc2[mt], false, false);
      }
    }
  }
  // branchless store (zeros into guard rows for p >= 50), transposed
#pragma unroll
  for (int mt = 0; mt < 2; ++mt)
#pragma unroll
    for (int r = 0; r < 8; ++r) {
      int M = mt * 16 + (laneHi ? 8 : 0) + r;
      float v = (p < 50) ? fmaxf(acc2[mt][r], 0.0f) : 0.0f;
      s_h2t[(p + 2) * 32 + M] = (_Float16)v;
    }
  __syncthreads();

  // ---- conv3: sum_kk W3[:,:,kk](16x32) @ h2[:, p+kk-2](32x50), WMMA ----
  float8 acc3;
#pragma unroll
  for (int r = 0; r < 8; ++r) acc3[r] = s_b3[(laneHi ? 8 : 0) + r];
#pragma unroll
  for (int kk = 0; kk < 5; ++kk) {
    const _Float16* brow = s_h2t + (p + kk) * 32;     // (p+kk-2)+2 pad rows
    half16 bfrag = frag_b_lds_contig(brow, 0, lane);
    half16 afrag = frag_a_lds(s_w3 + kk * 512, 32, 0, 0, lane);
    acc3 = __builtin_amdgcn_wmma_f32_16x16x32_f16(
        false, afrag, false, bfrag, (short)0, acc3, false, false);
  }
  // ReLU + mean over positions via LDS float atomics
  if (p < 50) {
#pragma unroll
    for (int r = 0; r < 8; ++r)
      atomicAdd(&s_sum[(laneHi ? 8 : 0) + r], fmaxf(acc3[r], 0.0f));
  }
  __syncthreads();
  if (tid < 16) feats[((size_t)(t * 16 + b)) * 16 + tid] = s_sum[tid] * (1.0f / 50.0f);
}

// ---------------------------------------------------------------------------
// Kernel 3/5: out[R x 256] = in[R x DIN] @ w(256 x DIN)^T + bias, WMMA tiles.
// DIN is compile-time so bound guards constant-fold. One tile per wave.
// ---------------------------------------------------------------------------
template <int DIN>
__global__ __launch_bounds__(128) void gemm_rows_x_wT(
    const float* __restrict__ in, const float* __restrict__ w,
    const float* __restrict__ bias, float* __restrict__ out) {
  const int lane = threadIdx.x & 31;
  const int wave = threadIdx.x >> 5;
  const int tile = blockIdx.x * 4 + wave;   // 32768 tiles total
  const int m0 = (tile >> 4) * 16;
  const int n0 = (tile & 15) * 16;
  const int laneHi = lane >> 4;
  const int lx = lane & 15;
  const int row = m0 + lx;
  const int col = n0 + lx;

  float8 acc;
#pragma unroll
  for (int r = 0; r < 8; ++r) acc[r] = bias[col];

  constexpr int NK = (DIN + 31) / 32;
#pragma unroll
  for (int ks = 0; ks < NK; ++ks) {
    const int kb = ks * 32;
    const int o1 = laneHi ? 16 : 0;
    const int o2 = laneHi ? 24 : 8;
    half16 afrag, bfrag;
#pragma unroll
    for (int j = 0; j < 8; ++j) {
      int k1 = kb + o1 + j, k2 = kb + o2 + j;
      afrag[j]     = (k1 < DIN) ? (_Float16)in[(size_t)row * DIN + k1] : (_Float16)0.0f;
      afrag[8 + j] = (k2 < DIN) ? (_Float16)in[(size_t)row * DIN + k2] : (_Float16)0.0f;
    }
#pragma unroll
    for (int j = 0; j < 16; ++j) {
      int k = kb + (laneHi ? 16 + j : j);
      bfrag[j] = (k < DIN) ? (_Float16)w[(size_t)col * DIN + k] : (_Float16)0.0f;
    }
    acc = __builtin_amdgcn_wmma_f32_16x16x32_f16(
        false, afrag, false, bfrag, (short)0, acc, false, false);
  }
#pragma unroll
  for (int r = 0; r < 8; ++r) {
    int M = m0 + (laneHi ? 8 : 0) + r;
    out[(size_t)M * 256 + n0 + lx] = acc[r];
  }
}

// ---------------------------------------------------------------------------
// Kernel 4/6: persistent single-workgroup LSTM recurrence.
// 16 waves x 16 gate columns; Whh^T B-fragments held in registers for all
// 2048 steps; h (f16) / c (f32) state and gate buffer in LDS.
// pre layout: [T][16][256], hs layout: [T][16][64].
// ---------------------------------------------------------------------------
__global__ __launch_bounds__(512) void lstm_scan(
    const float* __restrict__ pre, const float* __restrict__ whh,
    float* __restrict__ hs) {
  __shared__ __align__(16) _Float16 s_h[16 * 64];
  __shared__ float s_c[16 * 64];
  __shared__ float s_g[16 * 256];

  const int tid = threadIdx.x;
  const int lane = tid & 31;
  const int wave = tid >> 5;            // 0..15
  const int laneHi = lane >> 4;
  const int lx = lane & 15;
  const int n0 = wave * 16;
  const int col = n0 + lx;              // gate column 0..255

  // Preload B = Whh^T (64 x 256) fragments for this wave's column tile.
  half16 b0, b1;
#pragma unroll
  for (int j = 0; j < 16; ++j) {
    int k = laneHi ? 16 + j : j;
    b0[j] = (_Float16)whh[(size_t)col * 64 + k];
    b1[j] = (_Float16)whh[(size_t)col * 64 + 32 + k];
  }

  for (int i = tid; i < 16 * 64; i += 512) { s_h[i] = (_Float16)0.0f; s_c[i] = 0.0f; }
  __syncthreads();

  for (int t = 0; t < 2048; ++t) {
    const float* pt = pre + (size_t)t * 16 * 256;
    if (t + 1 < 2048)
      __builtin_prefetch(pre + (size_t)(t + 1) * 16 * 256 + tid * 8, 0, 0);

    half16 a0 = frag_a_lds(s_h, 64, 0, 0, lane);
    half16 a1 = frag_a_lds(s_h, 64, 0, 32, lane);
    float8 acc;
#pragma unroll
    for (int r = 0; r < 8; ++r)
      acc[r] = pt[(size_t)((laneHi ? 8 : 0) + r) * 256 + col];
    acc = __builtin_amdgcn_wmma_f32_16x16x32_f16(false, a0, false, b0, (short)0, acc, false, false);
    acc = __builtin_amdgcn_wmma_f32_16x16x32_f16(false, a1, false, b1, (short)0, acc, false, false);
#pragma unroll
    for (int r = 0; r < 8; ++r)
      s_g[((laneHi ? 8 : 0) + r) * 256 + col] = acc[r];
    __syncthreads();

    for (int e = tid; e < 16 * 64; e += 512) {
      int bb = e >> 6, j = e & 63;
      float gi = s_g[bb * 256 + j];
      float gf = s_g[bb * 256 + 64 + j];
      float gg = s_g[bb * 256 + 128 + j];
      float go = s_g[bb * 256 + 192 + j];
      float c = sigf(gf) * s_c[e] + sigf(gi) * tanhf(gg);
      float h = sigf(go) * tanhf(c);
      s_c[e] = c;
      s_h[e] = (_Float16)h;
      hs[(size_t)t * 16 * 64 + e] = h;
    }
    __syncthreads();
  }
}

// ---------------------------------------------------------------------------
// Kernel 7: FC head (64 -> 32 relu -> 2), one thread per (t,b) row.
// ---------------------------------------------------------------------------
__global__ __launch_bounds__(256) void fc_head(
    const float* __restrict__ hs, const float* __restrict__ fc1_w,
    const float* __restrict__ fc1_b, const float* __restrict__ fc2_w,
    const float* __restrict__ fc2_b, float* __restrict__ out) {
  int r = blockIdx.x * 256 + threadIdx.x;
  if (r >= 32768) return;
  int b = r & 15, t = r >> 4;           // hs rows are [t][b]
  const float* h = hs + (size_t)r * 64;
  float o0 = fc2_b[0], o1 = fc2_b[1];
  for (int j = 0; j < 32; ++j) {
    float z = fc1_b[j];
#pragma unroll
    for (int k = 0; k < 64; ++k) z += fc1_w[j * 64 + k] * h[k];
    z = fmaxf(z, 0.0f);
    o0 += fc2_w[j] * z;
    o1 += fc2_w[32 + j] * z;
  }
  out[((size_t)b * 2048 + t) * 2 + 0] = o0;
  out[((size_t)b * 2048 + t) * 2 + 1] = o1;
}

// ---------------------------------------------------------------------------
extern "C" void kernel_launch(void* const* d_in, const int* in_sizes, int n_in,
                              void* d_out, int out_size, void* d_ws, size_t ws_size,
                              hipStream_t stream) {
  (void)in_sizes; (void)n_in; (void)out_size; (void)ws_size;
  const float* x       = (const float*)d_in[0];
  const float* conv1_w = (const float*)d_in[1];
  const float* conv1_b = (const float*)d_in[2];
  const float* bn1_g   = (const float*)d_in[3];
  const float* bn1_b   = (const float*)d_in[4];
  const float* bn1_m   = (const float*)d_in[5];
  const float* bn1_v   = (const float*)d_in[6];
  const float* conv2_w = (const float*)d_in[7];
  const float* conv2_b = (const float*)d_in[8];
  const float* bn2_g   = (const float*)d_in[9];
  const float* bn2_b   = (const float*)d_in[10];
  const float* bn2_m   = (const float*)d_in[11];
  const float* bn2_v   = (const float*)d_in[12];
  const float* conv3_w = (const float*)d_in[13];
  const float* conv3_b = (const float*)d_in[14];
  const float* bn3_g   = (const float*)d_in[15];
  const float* bn3_b   = (const float*)d_in[16];
  const float* bn3_m   = (const float*)d_in[17];
  const float* bn3_v   = (const float*)d_in[18];
  const float* wih0    = (const float*)d_in[19];
  const float* whh0    = (const float*)d_in[20];
  const float* bih0    = (const float*)d_in[21];
  const float* bhh0    = (const float*)d_in[22];
  const float* wih1    = (const float*)d_in[23];
  const float* whh1    = (const float*)d_in[24];
  const float* bih1    = (const float*)d_in[25];
  const float* bhh1    = (const float*)d_in[26];
  const float* fc1_w   = (const float*)d_in[27];
  const float* fc1_b   = (const float*)d_in[28];
  const float* fc2_w   = (const float*)d_in[29];
  const float* fc2_b   = (const float*)d_in[30];

  char* ws = (char*)d_ws;
  float*    w1e   = (float*)(ws + 0);
  float*    b1e   = (float*)(ws + 4096);
  _Float16* w2e   = (_Float16*)(ws + 8192);        // 7*32*64 f16 = 28672 B
  float*    b2e   = (float*)(ws + 40960);
  _Float16* w3e   = (_Float16*)(ws + 45056);       // 5*16*32 f16 = 5120 B
  float*    b3e   = (float*)(ws + 53248);
  float*    bs0   = (float*)(ws + 57344);
  float*    bs1   = (float*)(ws + 61440);
  float*    feats = (float*)(ws + 65536);          // 32768*16 f32 = 2 MB
  float*    pre   = (float*)(ws + 2162688);        // 32768*256 f32 = 32 MB (reused)
  float*    hs0   = (float*)(ws + 35717120);       // 32768*64 f32 = 8 MB
  float*    hs1   = (float*)(ws + 44105728);       // 8 MB

  prep_weights<<<1, 256, 0, stream>>>(
      conv1_w, conv1_b, bn1_g, bn1_b, bn1_m, bn1_v,
      conv2_w, conv2_b, bn2_g, bn2_b, bn2_m, bn2_v,
      conv3_w, conv3_b, bn3_g, bn3_b, bn3_m, bn3_v,
      bih0, bhh0, bih1, bhh1,
      w1e, b1e, w2e, b2e, w3e, b3e, bs0, bs1);

  conv_fused<<<32768, 128, 0, stream>>>(x, w1e, b1e, w2e, b2e, w3e, b3e, feats);

  gemm_rows_x_wT<16><<<8192, 128, 0, stream>>>(feats, wih0, bs0, pre);
  lstm_scan<<<1, 512, 0, stream>>>(pre, whh0, hs0);
  gemm_rows_x_wT<64><<<8192, 128, 0, stream>>>(hs0, wih1, bs1, pre);
  lstm_scan<<<1, 512, 0, stream>>>(pre, whh1, hs1);

  fc_head<<<128, 256, 0, stream>>>(hs1, fc1_w, fc1_b, fc2_w, fc2_b, (float*)d_out);
}
